// Aresblock1_6_42064909697040
// MI455X (gfx1250) — 8-bit matrix operand order repaired and checked by reading, compiled, not run on hardware
//
#include <hip/hip_runtime.h>

typedef int   v16i __attribute__((ext_vector_type(16)));
typedef float v8f  __attribute__((ext_vector_type(8)));

#define BATCH 16
#define CFULL 256
#define CHALF 128
#define HH 56
#define WWID 56
#define PIX (HH*WWID)
#define GEPS 1e-5f

__device__ __forceinline__ unsigned int fp8sign(float v) {
  return (v > 0.f) ? 0x38u : ((v < 0.f) ? 0xB8u : 0x00u);
}

__global__ __launch_bounds__(256)
void k_prep1(const float* __restrict__ x, const float* __restrict__ move1,
             unsigned char* __restrict__ a1) {
  int t = blockIdx.x * 256 + threadIdx.x;
  if (t >= BATCH * PIX) return;
  int b = t / PIX, p = t % PIX;
  const float* xb = x + (size_t)b * CFULL * PIX + p;
  unsigned int* dst = (unsigned int*)(a1 + (size_t)t * CFULL);
  #pragma unroll 4
  for (int c4 = 0; c4 < CFULL / 4; ++c4) {
    unsigned int d = 0;
    #pragma unroll
    for (int k = 0; k < 4; ++k) {
      int c = c4 * 4 + k;
      int sc = (c & 1) * CHALF + (c >> 1);
      float v = xb[(size_t)sc * PIX] + move1[c];
      d |= fp8sign(v) << (8 * k);
    }
    dst[c4] = d;
  }
}

__global__ __launch_bounds__(288)
void k_wpack(const float* __restrict__ w, unsigned int* __restrict__ wp,
             float* __restrict__ sf) {
  int oc = blockIdx.x;
  int g = oc >> 6, ocl = oc & 63, nt = ocl >> 4, lidx = ocl & 15;
  const float* wb = w + (size_t)oc * 1152;
  __shared__ float red[288];
  float s = 0.f;
  for (int i = threadIdx.x; i < 1152; i += 288) s += fabsf(wb[i]);
  red[threadIdx.x] = s;
  __syncthreads();
  if (threadIdx.x == 0) {
    float tot = 0.f;
    for (int i = 0; i < 288; ++i) tot += red[i];
    sf[oc] = tot / 1152.f;
  }
  int tap = threadIdx.x / 32, cw = threadIdx.x % 32;
  int hi = cw >> 4, q = cw & 15;
  unsigned int d = 0;
  #pragma unroll
  for (int bb = 0; bb < 4; ++bb) {
    int ci = ((q >> 2) << 5) + (hi << 4) + ((q & 3) << 2) + bb;
    float v = wb[(size_t)ci * 9 + tap];
    d |= fp8sign(v) << (8 * bb);
  }
  wp[(size_t)((g * 4 + nt) * 9 + tap) * 512 + (size_t)(lidx + 16 * hi) * 16 + q] = d;
}

__global__ __launch_bounds__(256)
void k_conv_wmma(const unsigned char* __restrict__ act,
                 const unsigned int* __restrict__ wp,
                 const float* __restrict__ sf, const float* __restrict__ bias,
                 const float* __restrict__ pw, float* __restrict__ out) {
  const int MT = BATCH * PIX / 16;
  int wave = blockIdx.x * 8 + (threadIdx.x >> 5);
  int lane = threadIdx.x & 31;
  int g = wave / MT;
  int mt = wave % MT;
  int row = lane & 15, hi = lane >> 4;
  int pixr = mt * 16 + row;
  int b = pixr / PIX, p = pixr % PIX;
  int h = p / WWID, w = p % WWID;
  const unsigned char* abase = act + ((size_t)b * PIX) * CFULL + (size_t)g * CHALF;

  v8f acc[4];
  #pragma unroll
  for (int nt = 0; nt < 4; ++nt)
    #pragma unroll
    for (int r = 0; r < 8; ++r) acc[nt][r] = 0.f;

  #pragma unroll
  for (int tap = 0; tap < 9; ++tap) {
    int dh = tap / 3 - 1, dw = tap % 3 - 1;
    int hh = h + dh, ww2 = w + dw;
    bool ok = (hh >= 0) & (hh < HH) & (ww2 >= 0) & (ww2 < WWID);
    const unsigned char* ap = abase + (size_t)(hh * WWID + ww2) * CFULL;
    v16i A;
    #pragma unroll
    for (int j = 0; j < 8; ++j) {
      int off = ((j >> 1) << 5) + (hi << 4) + ((j & 1) << 3);
      unsigned long long d = ok ? *(const unsigned long long*)(ap + off) : 0ull;
      A[(j >> 2) * 8 + (j & 3) * 2 + 0] = (int)(unsigned int)(d & 0xffffffffull);
      A[(j >> 2) * 8 + (j & 3) * 2 + 1] = (int)(unsigned int)(d >> 32);
    }
    #pragma unroll
    for (int nt = 0; nt < 4; ++nt) {
      const int* wb = (const int*)wp + (size_t)((g * 4 + nt) * 9 + tap) * 512 +
                      (size_t)lane * 16;
      v16i Bf;
      #pragma unroll
      for (int q = 0; q < 16; ++q) Bf[q] = wb[q];
      acc[nt] = __builtin_amdgcn_wmma_f32_16x16x128_fp8_fp8(
          A, Bf, (short)0, acc[nt], false, false);
    }
  }

  int p0 = (mt % (PIX / 16)) * 16 + hi * 8;
  #pragma unroll
  for (int nt = 0; nt < 4; ++nt) {
    int oc = g * 64 + nt * 16 + (lane & 15);
    float s = sf[oc], bi = bias[oc], pv = pw[oc];
    float* op = out + ((size_t)b * CHALF + oc) * PIX + p0;
    float4 o0, o1;
    float t0;
    t0 = acc[nt][0] * s + bi; o0.x = t0 >= 0.f ? t0 : t0 * pv;
    t0 = acc[nt][1] * s + bi; o0.y = t0 >= 0.f ? t0 : t0 * pv;
    t0 = acc[nt][2] * s + bi; o0.z = t0 >= 0.f ? t0 : t0 * pv;
    t0 = acc[nt][3] * s + bi; o0.w = t0 >= 0.f ? t0 : t0 * pv;
    t0 = acc[nt][4] * s + bi; o1.x = t0 >= 0.f ? t0 : t0 * pv;
    t0 = acc[nt][5] * s + bi; o1.y = t0 >= 0.f ? t0 : t0 * pv;
    t0 = acc[nt][6] * s + bi; o1.z = t0 >= 0.f ? t0 : t0 * pv;
    t0 = acc[nt][7] * s + bi; o1.w = t0 >= 0.f ? t0 : t0 * pv;
    *(float4*)(op) = o0;
    *(float4*)(op + 4) = o1;
  }
}

__global__ __launch_bounds__(256)
void k_gn_reduce(const float* __restrict__ y, float* __restrict__ st) {
  int bg = blockIdx.x;
  const float* base = y + (size_t)bg * 64 * PIX;
  __shared__ float s1[256], s2[256];
  float a = 0.f, b2 = 0.f;
  const int N = 64 * PIX;
  for (int i = threadIdx.x; i < N; i += 256) {
    float v = base[i]; a += v; b2 += v * v;
  }
  s1[threadIdx.x] = a; s2[threadIdx.x] = b2;
  __syncthreads();
  for (int stp = 128; stp > 0; stp >>= 1) {
    if (threadIdx.x < stp) {
      s1[threadIdx.x] += s1[threadIdx.x + stp];
      s2[threadIdx.x] += s2[threadIdx.x + stp];
    }
    __syncthreads();
  }
  if (threadIdx.x == 0) {
    float m = s1[0] / N, var = s2[0] / N - m * m;
    st[bg * 2] = m;
    st[bg * 2 + 1] = rsqrtf(var + GEPS);
  }
}

__global__ __launch_bounds__(256)
void k_gn_apply(float* __restrict__ y, const float* __restrict__ st,
                const float* __restrict__ gg, const float* __restrict__ gb,
                const float* __restrict__ ab, const float* __restrict__ pp) {
  size_t total = (size_t)BATCH * CHALF * PIX;
  for (size_t i = blockIdx.x * 256ull + threadIdx.x; i < total;
       i += (size_t)gridDim.x * 256ull) {
    int c = (int)((i / PIX) % CHALF);
    int b = (int)(i / ((size_t)CHALF * PIX));
    int g = c >> 6;
    float m = st[(b * 2 + g) * 2], r = st[(b * 2 + g) * 2 + 1];
    float v = (y[i] - m) * r * gg[c] + gb[c];
    v += ab[c];
    v = v >= 0.f ? v : v * pp[c];
    y[i] = v;
  }
}

__global__ __launch_bounds__(256)
void k_bn_reduce(const float* __restrict__ src, int C,
                 const float* __restrict__ gamma, const float* __restrict__ beta,
                 float* __restrict__ st) {
  int c = blockIdx.x;
  __shared__ float s1[256], s2[256];
  float a = 0.f, b2 = 0.f;
  for (int bp = threadIdx.x; bp < BATCH * PIX; bp += 256) {
    int b = bp / PIX, p = bp % PIX;
    float v = src[((size_t)b * C + c) * PIX + p];
    a += v; b2 += v * v;
  }
  s1[threadIdx.x] = a; s2[threadIdx.x] = b2;
  __syncthreads();
  for (int stp = 128; stp > 0; stp >>= 1) {
    if (threadIdx.x < stp) {
      s1[threadIdx.x] += s1[threadIdx.x + stp];
      s2[threadIdx.x] += s2[threadIdx.x + stp];
    }
    __syncthreads();
  }
  if (threadIdx.x == 0) {
    float n = (float)(BATCH * PIX);
    float m = s1[0] / n, var = s2[0] / n - m * m;
    float sc = gamma[c] * rsqrtf(var + GEPS);
    st[c * 2] = sc;
    st[c * 2 + 1] = beta[c] - m * sc;
  }
}

__global__ __launch_bounds__(256)
void k_mid(const float* __restrict__ x, const float* __restrict__ t1,
           const float* __restrict__ bn1st, const float* __restrict__ move21,
           const float* __restrict__ p2, const float* __restrict__ move22,
           const float* __restrict__ move31, float* __restrict__ x2s,
           unsigned char* __restrict__ a2) {
  int t = blockIdx.x * 256 + threadIdx.x;
  if (t >= BATCH * PIX) return;
  int b = t / PIX, p = t % PIX;
  const float* xb = x + (size_t)b * CFULL * PIX + p;
  const float* t1b = t1 + (size_t)b * CHALF * PIX + p;
  float* x2b = x2s + (size_t)b * CFULL * PIX + p;
  unsigned int* a2d = (unsigned int*)(a2 + (size_t)t * CFULL);
  #pragma unroll 4
  for (int c4 = 0; c4 < CFULL / 4; ++c4) {
    unsigned int dd = 0;
    #pragma unroll
    for (int k = 0; k < 4; ++k) {
      int c2 = c4 * 4 + k;
      int cu = (c2 & 1) * CHALF + (c2 >> 1);
      int sc = (cu & 1) * CHALF + (cu >> 1);
      float xs = xb[(size_t)sc * PIX];
      float u;
      if (cu < CHALF)
        u = bn1st[cu * 2] * t1b[(size_t)cu * PIX] + bn1st[cu * 2 + 1] + xs;
      else
        u = xs;
      u += move21[cu];
      u = u >= 0.f ? u : u * p2[cu];
      u += move22[cu];
      x2b[(size_t)c2 * PIX] = u;
      dd |= fp8sign(u + move31[c2]) << (8 * k);
    }
    a2d[c4] = dd;
  }
}

__global__ __launch_bounds__(256)
void k_final(const float* __restrict__ x, const float* __restrict__ t3,
             const float* __restrict__ bn3st, const float* __restrict__ move41,
             const float* __restrict__ p4, const float* __restrict__ move42,
             float* __restrict__ x2s) {
  int t = blockIdx.x * 256 + threadIdx.x;
  if (t >= BATCH * PIX) return;
  int b = t / PIX, p = t % PIX;
  const float* xb = x + (size_t)b * CFULL * PIX + p;
  const float* t3b = t3 + (size_t)b * CHALF * PIX + p;
  float* xo = x2s + (size_t)b * CFULL * PIX + p;
  for (int c = 0; c < CFULL; ++c) {
    float v = xo[(size_t)c * PIX];
    if (c < CHALF)
      v = bn3st[c * 2] * t3b[(size_t)c * PIX] + bn3st[c * 2 + 1] + v;
    v += move41[c];
    v = v >= 0.f ? v : v * p4[c];
    v += move42[c];
    v += xb[(size_t)c * PIX];
    xo[(size_t)c * PIX] = v;
  }
}

__global__ __launch_bounds__(256)
void k_bn_apply(const float* __restrict__ src, const float* __restrict__ st,
                float* __restrict__ out) {
  size_t total = (size_t)BATCH * CFULL * PIX;
  for (size_t i = blockIdx.x * 256ull + threadIdx.x; i < total;
       i += (size_t)gridDim.x * 256ull) {
    int c = (int)((i / PIX) % CFULL);
    out[i] = st[c * 2] * src[i] + st[c * 2 + 1];
  }
}

extern "C" void kernel_launch(void* const* d_in, const int* in_sizes, int n_in,
                              void* d_out, int out_size, void* d_ws, size_t ws_size,
                              hipStream_t stream) {
  (void)in_sizes; (void)n_in; (void)out_size; (void)ws_size;
  const float* x      = (const float*)d_in[0];
  const float* w3     = (const float*)d_in[1];
  const float* b3     = (const float*)d_in[2];
  const float* pw3    = (const float*)d_in[3];
  const float* gg3    = (const float*)d_in[4];
  const float* gb3    = (const float*)d_in[5];
  const float* w1     = (const float*)d_in[6];
  const float* b1     = (const float*)d_in[7];
  const float* pw1    = (const float*)d_in[8];
  const float* gg1    = (const float*)d_in[9];
  const float* gb1    = (const float*)d_in[10];
  const float* move1  = (const float*)d_in[11];
  const float* ab1    = (const float*)d_in[12];
  const float* p1     = (const float*)d_in[13];
  const float* bn1g   = (const float*)d_in[14];
  const float* bn1b   = (const float*)d_in[15];
  const float* move21 = (const float*)d_in[16];
  const float* p2     = (const float*)d_in[17];
  const float* move22 = (const float*)d_in[18];
  const float* move31 = (const float*)d_in[19];
  const float* ab2    = (const float*)d_in[20];
  const float* p3     = (const float*)d_in[21];
  const float* bn3g   = (const float*)d_in[22];
  const float* bn3b   = (const float*)d_in[23];
  const float* move41 = (const float*)d_in[24];
  const float* p4     = (const float*)d_in[25];
  const float* move42 = (const float*)d_in[26];
  const float* bng    = (const float*)d_in[27];
  const float* bnb    = (const float*)d_in[28];
  float* out = (float*)d_out;

  char* ws = (char*)d_ws;
  unsigned char* aAct = (unsigned char*)(ws + 0);
  float*        yBuf  = (float*)(ws + 12845056);
  float*        x2s   = (float*)(ws + 38535168);
  unsigned int* wp    = (unsigned int*)(ws + 89915392);
  float*        sf    = (float*)(ws + 90062848);
  float*        gnst  = (float*)(ws + 90063360);
  float*        bn1st = (float*)(ws + 90063616);
  float*        bn3st = (float*)(ws + 90064640);
  float*        bnFst = (float*)(ws + 90065664);

  const int nBP = BATCH * PIX;
  dim3 blk(256);

  k_prep1<<<nBP / 256, blk, 0, stream>>>(x, move1, aAct);
  k_wpack<<<128, 288, 0, stream>>>(w3, wp, sf);
  k_conv_wmma<<<784, blk, 0, stream>>>(aAct, wp, sf, b3, pw3, yBuf);
  k_gn_reduce<<<32, blk, 0, stream>>>(yBuf, gnst);
  k_gn_apply<<<3136, blk, 0, stream>>>(yBuf, gnst, gg3, gb3, ab1, p1);
  k_bn_reduce<<<128, blk, 0, stream>>>(yBuf, CHALF, bn1g, bn1b, bn1st);
  k_mid<<<nBP / 256, blk, 0, stream>>>(x, yBuf, bn1st, move21, p2, move22,
                                       move31, x2s, aAct);

  k_wpack<<<128, 288, 0, stream>>>(w1, wp, sf);
  k_conv_wmma<<<784, blk, 0, stream>>>(aAct, wp, sf, b1, pw1, yBuf);
  k_gn_reduce<<<32, blk, 0, stream>>>(yBuf, gnst);
  k_gn_apply<<<3136, blk, 0, stream>>>(yBuf, gnst, gg1, gb1, ab2, p3);
  k_bn_reduce<<<128, blk, 0, stream>>>(yBuf, CHALF, bn3g, bn3b, bn3st);
  k_final<<<nBP / 256, blk, 0, stream>>>(x, yBuf, bn3st, move41, p4, move42, x2s);

  k_bn_reduce<<<256, blk, 0, stream>>>(x2s, CFULL, bng, bnb, bnFst);
  k_bn_apply<<<3136, blk, 0, stream>>>(x2s, bnFst, out);
}
